// MiMoV2MoEGate_7172595384906
// MI455X (gfx1250) — compile-verified, matrix operand order checked by reading, not run on hardware
//
#include <hip/hip_runtime.h>
#include <hip/hip_bf16.h>

typedef __attribute__((ext_vector_type(16))) __bf16 v16bf;
typedef __attribute__((ext_vector_type(8)))  float  v8f;
typedef __attribute__((ext_vector_type(4)))  unsigned int u32x4;
typedef __attribute__((ext_vector_type(8)))  int i32x8;
typedef __attribute__((ext_vector_type(4)))  int i32x4;

#define HDIM   4096
#define NEXP   256
#define TOPK   8
#define TOPKG  4
#define MBLK   64
#define KT     64
#define LDP    72      // padded LDS row stride (bf16 elements) = 144 B
#define NEGINF (-1e30f)

union Frag {
    v16bf v;
    uint4 q[2];
};

// Shared: GEMM staging tiles aliased with the score tile (disjoint phases).
struct SharedGemm {
    unsigned short A[MBLK][LDP];        //  9216 B (manual f32->bf16 staging)
    unsigned short B[2][NEXP][LDP];     // 73728 B (TDM double buffer, hw-padded rows)
};
union SharedMem {
    SharedGemm g;                       // 82944 B
    float S[MBLK][NEXP];                // 65536 B
};

__device__ __forceinline__ unsigned int f32_to_bf16(float f) {
    unsigned int u = __float_as_uint(f);
    u += 0x7FFFu + ((u >> 16) & 1u);    // round-to-nearest-even
    return u >> 16;
}

// ---- Tensor Data Mover: 2D tile load (NEXP rows x KT bf16) global -> LDS ----
// D# per CDNA5 ISA ch.8: group0 {count, lds_addr, global_addr, type=2},
// group1 {data_size=2B, pad_enable, pad_interval=32dw (one 128B row),
//         pad_amount=4dw (16B) -> LDS row stride 144B = LDP,
//         tensor_dim0=HDIM, tensor_dim1=NEXP, tile=KT x NEXP, dim0_stride=HDIM}
__device__ __forceinline__ void tdm_load_b_tile(const unsigned short* gsrc,
                                                unsigned int lds_off) {
    unsigned long long ga = (unsigned long long)(const void*)gsrc;
    u32x4 g0;
    g0[0] = 1u;                                         // count = 1 valid descriptor
    g0[1] = lds_off;                                    // LDS byte address
    g0[2] = (unsigned int)(ga & 0xFFFFFFFFu);           // global_addr[31:0]
    g0[3] = (unsigned int)((ga >> 32) & 0x1FFFFFFu)     // global_addr[56:32]
          | (2u << 30);                                 // type = 2 (image)
    i32x8 g1;
    g1[0] = (1 << 16)                                   // data_size = 2 bytes
          | (1 << 20)                                   // pad_enable
          | (4 << 22)                                   // pad_interval: 32 DWORDs
          | (3 << 25);                                  // pad_amount: 4 DWORDs
    g1[1] = (HDIM & 0xFFFF) << 16;                      // tensor_dim0[15:0]
    g1[2] = ((HDIM >> 16) & 0xFFFF)                     // tensor_dim0[31:16]
          | ((NEXP & 0xFFFF) << 16);                    // tensor_dim1[15:0]
    g1[3] = ((NEXP >> 16) & 0xFFFF)                     // tensor_dim1[31:16]
          | (KT << 16);                                 // tile_dim0 = 64
    g1[4] = NEXP;                                       // tile_dim1 = 256, tile_dim2 = 0
    g1[5] = HDIM;                                       // tensor_dim0_stride[31:0]
    g1[6] = 0;                                          // stride hi / dim1_stride
    g1[7] = 0;
    i32x4 z4 = {0, 0, 0, 0};
    i32x8 z8 = {0, 0, 0, 0, 0, 0, 0, 0};
    // 6-arg form (clang-23 / therock headers): groups 0-3, aux group, cpol
    __builtin_amdgcn_tensor_load_to_lds(g0, g1, z4, z4, z8, 0);
}

// ---- pre-pass: convert gate weights f32 -> bf16 once (4MB -> 2MB, L2-resident) ----
__global__ __launch_bounds__(256)
void convert_w_bf16(const float* __restrict__ w, unsigned int* __restrict__ out) {
    const int i = blockIdx.x * blockDim.x + threadIdx.x;     // 8 floats per thread
    const float4* src = (const float4*)w;
    float4 a = src[i * 2 + 0];
    float4 b = src[i * 2 + 1];
    uint4 o;
    o.x = f32_to_bf16(a.x) | (f32_to_bf16(a.y) << 16);
    o.y = f32_to_bf16(a.z) | (f32_to_bf16(a.w) << 16);
    o.z = f32_to_bf16(b.x) | (f32_to_bf16(b.y) << 16);
    o.w = f32_to_bf16(b.z) | (f32_to_bf16(b.w) << 16);
    ((uint4*)out)[i] = o;
}

__global__ __launch_bounds__(256)
void moe_gate_fused(const float* __restrict__ x,
                    const unsigned short* __restrict__ wbf,   // bf16 weights [NEXP][HDIM]
                    const float* __restrict__ bias,
                    int*   __restrict__ outIdx,
                    float* __restrict__ outW,
                    int T) {
    __shared__ SharedMem sh;

    const int tid   = threadIdx.x;
    const int wave  = tid >> 5;
    const int lane  = tid & 31;
    const int half  = lane >> 4;     // 0: lanes 0-15, 1: lanes 16-31
    const int r     = lane & 15;
    const int waveM = wave >> 2;     // 0..1 -> 32-row slice
    const int waveN = wave & 3;      // 0..3 -> 64-col slice
    const int mBase = blockIdx.x * MBLK;

    v8f zero = {};
    v8f acc[2][4];
#pragma unroll
    for (int mt = 0; mt < 2; ++mt)
#pragma unroll
        for (int nt = 0; nt < 4; ++nt) acc[mt][nt] = zero;

    const unsigned int ldsB0 = (unsigned int)(unsigned long long)(void*)&sh.g.B[0][0][0];
    const unsigned int ldsB1 = (unsigned int)(unsigned long long)(void*)&sh.g.B[1][0][0];

    // Preload B tile for stage 0 via TDM (wave 0 issues; EXEC-independent DMA)
    if (wave == 0) {
        tdm_load_b_tile(wbf, ldsB0);
    }

    // ---------------- GEMM: logits = x . W^T (bf16 WMMA, f32 accum) ----------------
    int cur = 0;
    for (int kt = 0; kt < HDIM; kt += KT) {
        // Stage x: global f32 -> registers (64 rows x 16 float4, 4 per thread)
        float4 aReg[4];
#pragma unroll
        for (int i = 0; i < 4; ++i) {
            int id = tid + i * 256;
            int row = id >> 4, c4 = id & 15;
            aReg[i] = *(const float4*)(x + (size_t)(mBase + row) * HDIM + kt + c4 * 4);
        }

        __syncthreads();   // previous stage's LDS reads (A and prev B buffer) complete

        // Convert f32 -> bf16 (RNE) and store x tile to LDS
#pragma unroll
        for (int i = 0; i < 4; ++i) {
            int id = tid + i * 256;
            int row = id >> 4, c4 = id & 15;
            unsigned int p0 = f32_to_bf16(aReg[i].x) | (f32_to_bf16(aReg[i].y) << 16);
            unsigned int p1 = f32_to_bf16(aReg[i].z) | (f32_to_bf16(aReg[i].w) << 16);
            *(uint2*)(&sh.g.A[row][c4 * 4]) = make_uint2(p0, p1);
        }

        // Wave 0: kick TDM for next stage's B tile, then ensure current tile landed
        if (wave == 0) {
            if (kt + KT < HDIM) {
                tdm_load_b_tile(wbf + (kt + KT), (cur ^ 1) ? ldsB1 : ldsB0);
                __builtin_amdgcn_s_wait_tensorcnt(1);   // all but newest complete
            } else {
                __builtin_amdgcn_s_wait_tensorcnt(0);
            }
        }

        // Prefetch next K-stage of x into caches
        if (kt + KT < HDIM) {
#pragma unroll
            for (int i = 0; i < 4; ++i) {
                int id = tid + i * 256;
                int row = id >> 4, c4 = id & 15;
                __builtin_prefetch(x + (size_t)(mBase + row) * HDIM + kt + KT + c4 * 4, 0, 1);
            }
        }

        __syncthreads();   // A stored, current B tile complete -> compute

        // Two K=32 WMMA steps per stage
#pragma unroll
        for (int k2 = 0; k2 < 2; ++k2) {
            const int kb = k2 * 32 + half * 8;  // lanes 0-15: K{0..7,16..23}; 16-31: K{8..15,24..31}
            Frag a[2];
#pragma unroll
            for (int mt = 0; mt < 2; ++mt) {
                int row = waveM * 32 + mt * 16 + r;
                a[mt].q[0] = *(const uint4*)(&sh.g.A[row][kb]);
                a[mt].q[1] = *(const uint4*)(&sh.g.A[row][kb + 16]);
            }
            Frag b[4];
#pragma unroll
            for (int nt = 0; nt < 4; ++nt) {
                int row = waveN * 64 + nt * 16 + r;
                b[nt].q[0] = *(const uint4*)(&sh.g.B[cur][row][kb]);
                b[nt].q[1] = *(const uint4*)(&sh.g.B[cur][row][kb + 16]);
            }
#pragma unroll
            for (int mt = 0; mt < 2; ++mt)
#pragma unroll
                for (int nt = 0; nt < 4; ++nt)
                    acc[mt][nt] = __builtin_amdgcn_wmma_f32_16x16x32_bf16(
                        false, a[mt].v, false, b[nt].v,
                        (short)0, acc[mt][nt], false, false);
        }
        cur ^= 1;
    }

    __syncthreads();   // all waves done with sh.g before aliasing as sh.S

    // ---------------- Epilogue: sigmoid -> LDS score tile ----------------
#pragma unroll
    for (int mt = 0; mt < 2; ++mt)
#pragma unroll
        for (int nt = 0; nt < 4; ++nt)
#pragma unroll
            for (int i = 0; i < 8; ++i) {
                int rowL = waveM * 32 + mt * 16 + half * 8 + i;
                int colL = waveN * 64 + nt * 16 + r;
                float v = acc[mt][nt][i];
                sh.S[rowL][colL] = 1.0f / (1.0f + __expf(-v));
            }
    __syncthreads();

    // ---------------- Routing: one wave per token, 8 tokens per wave ----------------
    float bb[8];
    {
        float4 b0 = *(const float4*)(bias + lane * 8);
        float4 b1 = *(const float4*)(bias + lane * 8 + 4);
        bb[0] = b0.x; bb[1] = b0.y; bb[2] = b0.z; bb[3] = b0.w;
        bb[4] = b1.x; bb[5] = b1.y; bb[6] = b1.z; bb[7] = b1.w;
    }

    for (int tok = 0; tok < 8; ++tok) {
        const int tl = wave * 8 + tok;
        const int t  = mBase + tl;
        if (t >= T) break;

        float sraw[8];
#pragma unroll
        for (int i = 0; i < 8; ++i) sraw[i] = sh.S[tl][lane * 8 + i];

        // group score = sum of top-2 of scores_for_choice within each 32-expert group
        float m1 = NEGINF, m2 = NEGINF;
#pragma unroll
        for (int i = 0; i < 8; ++i) {
            float v = sraw[i] + bb[i];
            if (v > m1) { m2 = m1; m1 = v; }
            else if (v > m2) { m2 = v; }
        }
#pragma unroll
        for (int off = 1; off <= 2; off <<= 1) {  // merge top-2 across the 4 lanes of a group
            float o1 = __shfl_xor(m1, off, 32);
            float o2 = __shfl_xor(m2, off, 32);
            float n1 = fmaxf(m1, o1);
            float n2 = fmaxf(fminf(m1, o1), fmaxf(m2, o2));
            m1 = n1; m2 = n2;
        }
        const float gs = m1 + m2;
        const int   g  = lane >> 2;
        int rank = 0;
#pragma unroll
        for (int j = 0; j < 8; ++j) {
            float gj = __shfl(gs, j * 4, 32);
            rank += (gj > gs) || (gj == gs && j < g);
        }
        const bool keep = rank < TOPKG;

        float cand[8];
#pragma unroll
        for (int i = 0; i < 8; ++i) cand[i] = keep ? (sraw[i] + bb[i]) : NEGINF;

        float sum = 0.0f, myW = 0.0f;
        int myIdx = 0;
#pragma unroll
        for (int it = 0; it < TOPK; ++it) {
            float lv = cand[0];
            int   li = lane * 8;
#pragma unroll
            for (int i = 1; i < 8; ++i)
                if (cand[i] > lv) { lv = cand[i]; li = lane * 8 + i; }
#pragma unroll
            for (int off = 16; off >= 1; off >>= 1) {
                float ov = __shfl_xor(lv, off, 32);
                int   oi = __shfl_xor(li, off, 32);
                if (ov > lv || (ov == lv && oi < li)) { lv = ov; li = oi; }
            }
            const int owner = li >> 3, slot = li & 7;
            float mine = sraw[0];
#pragma unroll
            for (int i = 1; i < 8; ++i) if (slot == i) mine = sraw[i];
            float wsel = __shfl(mine, owner, 32);   // weight from raw sigmoid score
            if (lane == owner) {
#pragma unroll
                for (int i = 0; i < 8; ++i) if (slot == i) cand[i] = NEGINF;
            }
            if (lane == it) { myIdx = li; myW = wsel; }
            sum += wsel;
        }

        const float scale = 2.5f / (sum + 1e-20f);
        if (lane < TOPK) {
            outIdx[(size_t)t * TOPK + lane] = myIdx;
            outW[(size_t)t * TOPK + lane]   = myW * scale;
        }
    }
}

extern "C" void kernel_launch(void* const* d_in, const int* in_sizes, int n_in,
                              void* d_out, int out_size, void* d_ws, size_t ws_size,
                              hipStream_t stream) {
    const float* x    = (const float*)d_in[0];  // [4,4096,4096] f32
    const float* w    = (const float*)d_in[1];  // [256,4096]    f32
    const float* bias = (const float*)d_in[2];  // [256]         f32
    const int T = in_sizes[0] / HDIM;           // 16384 tokens

    int*   outIdx = (int*)d_out;                        // first T*8 elements: indices
    float* outW   = (float*)d_out + (size_t)T * TOPK;   // next  T*8 elements: weights

    unsigned int* wbf = (unsigned int*)d_ws;            // bf16 weights (2 MB) in scratch

    hipLaunchKernelGGL(convert_w_bf16, dim3((NEXP * HDIM) / (256 * 8)), dim3(256),
                       0, stream, w, wbf);
    hipLaunchKernelGGL(moe_gate_fused, dim3(T / MBLK), dim3(256), 0, stream,
                       x, (const unsigned short*)wbf, bias, outIdx, outW, T);
}
